// MultiheadSelfAttention_32409823216354
// MI455X (gfx1250) — matrix operand order repaired and checked by reading, not compiled, not run on hardware
//
#include <hip/hip_runtime.h>
#include <hip/hip_bf16.h>

typedef __bf16 bf16_t;
typedef bf16_t v16bf __attribute__((ext_vector_type(16)));
typedef bf16_t v8bf  __attribute__((ext_vector_type(8)));
typedef float  v8f   __attribute__((ext_vector_type(8)));

// Problem constants (match reference setup_inputs)
#define BB 2
#define TT 2048
#define MM 1024
#define HH 8
#define DD 128
#define HD (HH*DD)      // 1024

__device__ __forceinline__ v8bf ld8(const bf16_t* p) {
    return *reinterpret_cast<const v8bf*>(p);
}
__device__ __forceinline__ v16bf cat16(v8bf a, v8bf b) {
    return __builtin_shufflevector(a, b, 0,1,2,3,4,5,6,7,8,9,10,11,12,13,14,15);
}

// A fragment 16x32 (MxK): element (m,k) at p[(mBase+m)*ld + kBase+k]
// lanes 0-15: row=lane, K {0..7, 16..23}; lanes 16-31: row=lane-16, K {8..15, 24..31}
__device__ __forceinline__ v16bf load_afrag(const bf16_t* p, int ld, int mBase, int kBase, int lane) {
    int m  = mBase + (lane & 15);
    int k0 = kBase + ((lane >> 4) << 3);          // +0 or +8
    const bf16_t* row = p + m * ld;
    return cat16(ld8(row + k0), ld8(row + k0 + 16));
}

// B fragment 32x16 (KxN) stored transposed: element (k,n) at p[(nBase+n)*ld + kBase+k]
// lanes 0-15: col=lane, K {0..7, 16..23}; lanes 16-31: col=lane-16, K {8..15, 24..31}
__device__ __forceinline__ v16bf load_bfrag(const bf16_t* p, int ld, int nBase, int kBase, int lane) {
    int n  = nBase + (lane & 15);
    int k0 = kBase + ((lane >> 4) << 3);          // +0 or +8
    const bf16_t* row = p + n * ld;
    return cat16(ld8(row + k0), ld8(row + k0 + 16));
}

__device__ __forceinline__ v8f wmma_bf16(v16bf a, v16bf b, v8f c) {
    return __builtin_amdgcn_wmma_f32_16x16x32_bf16(false, a, false, b, (short)0, c, false, false);
}

// CDNA5 async global->LDS copy (ASYNCcnt-tracked, bypasses VGPRs).
// Flat->LDS mapping uses addr[31:0], so truncating the generic shared pointer
// yields the LDS byte address for the VDST operand.
__device__ __forceinline__ void async_ld_b128(void* lds, const void* g) {
    unsigned lds32 = (unsigned)(unsigned long long)lds;
    asm volatile("global_load_async_to_lds_b128 %0, %1, off"
                 :: "v"(lds32), "v"(g) : "memory");
}
#define WAIT_ASYNC(n) asm volatile("s_wait_asynccnt " #n ::: "memory")

// ---------------------------------------------------------------- converts
__global__ __launch_bounds__(256) void cvt_f32_bf16(const float* __restrict__ src,
                                                    bf16_t* __restrict__ dst, int n) {
    int i = blockIdx.x * 256 + threadIdx.x;
    if (i < n) dst[i] = (bf16_t)src[i];
}

// src f32 [K][N] row-major  ->  dst bf16 [N][K] (transposed), K==N==1024 here
__global__ __launch_bounds__(256) void cvt_transpose_bf16(const float* __restrict__ src,
                                                          bf16_t* __restrict__ dst,
                                                          int K, int N) {
    int i = blockIdx.x * 256 + threadIdx.x;
    if (i < K * N) {
        int k = i / N, n = i % N;
        dst[(size_t)n * K + k] = (bf16_t)src[i];
    }
}

// ---------------------------------------------------------------- QKV GEMM + fused RoPE
// grid: (T/64, H, B*3)  block: 256 (8 waves)
// WG tile: 64 rows (time) x 128 cols (one head's full D), K = M = 1024.
// A and B^T tiles stream via async-to-LDS, double buffered (3 async instr/batch).
__global__ __launch_bounds__(256) void qkv_rope_kernel(
    const bf16_t* __restrict__ xb,                  // [B*T][M]
    const bf16_t* __restrict__ wqt,
    const bf16_t* __restrict__ wkt,
    const bf16_t* __restrict__ wvt,                 // transposed: [H*D][M]
    bf16_t* __restrict__ qb, bf16_t* __restrict__ kb,  // [B][H][T][D]
    bf16_t* __restrict__ vb)                           // V transposed: [B][H][D][T]
{
    __shared__ bf16_t Alds[2][64 * 32];    // 2 x 4 KB
    __shared__ bf16_t Btl [2][128 * 32];   // 2 x 8 KB   (B^T tile: [n][k])
    __shared__ float  Cst [64 * 128];      // 32 KB fp32 stage for epilogue

    const int tq  = blockIdx.x;
    const int h   = blockIdx.y;
    const int bz  = blockIdx.z;
    const int b   = bz / 3, sel = bz % 3;
    const bf16_t* wt = (sel == 0) ? wqt : (sel == 1) ? wkt : wvt;

    const int row0 = tq * 64;
    const bf16_t* Arow = xb + (size_t)(b * TT + row0) * MM;
    const bf16_t* Wrow = wt + (size_t)(h * DD) * MM;   // rows n=h*128.., ld=M

    const int tid  = threadIdx.x;
    const int wid  = tid >> 5, lane = tid & 31;
    const int wm   = wid & 3,  wn   = wid >> 2;
    const int ar   = tid >> 2, akc = tid & 3;          // A-tile chunk for this thread
    const int brow0 = tid >> 2, bkc = tid & 3;         // B-tile chunk (i*64 rows apart)

    v8f acc[4] = {};

    // prologue: batch 0
    async_ld_b128(&Alds[0][ar * 32 + akc * 8], Arow + ar * MM + akc * 8);
    async_ld_b128(&Btl[0][brow0 * 32 + bkc * 8], Wrow + (size_t)brow0 * MM + bkc * 8);
    async_ld_b128(&Btl[0][(brow0 + 64) * 32 + bkc * 8], Wrow + (size_t)(brow0 + 64) * MM + bkc * 8);

    for (int k0 = 0; k0 < MM; k0 += 32) {
        const int cur = (k0 >> 5) & 1;
        __syncthreads();                       // everyone done computing buf[cur^1]
        if (k0 + 32 < MM) {
            const int nxt = cur ^ 1, kn = k0 + 32;
            async_ld_b128(&Alds[nxt][ar * 32 + akc * 8], Arow + ar * MM + kn + akc * 8);
            async_ld_b128(&Btl[nxt][brow0 * 32 + bkc * 8], Wrow + (size_t)brow0 * MM + kn + bkc * 8);
            async_ld_b128(&Btl[nxt][(brow0 + 64) * 32 + bkc * 8], Wrow + (size_t)(brow0 + 64) * MM + kn + bkc * 8);
            WAIT_ASYNC(3);                     // in-order: batch 'cur' has landed
        } else {
            WAIT_ASYNC(0);
        }
        __syncthreads();                       // all waves' batch 'cur' complete
        v16bf af = load_afrag(Alds[cur], 32, wm * 16, 0, lane);
        #pragma unroll
        for (int tn = 0; tn < 4; tn++) {
            v16bf bfg = load_bfrag(Btl[cur], 32, wn * 64 + tn * 16, 0, lane);
            acc[tn] = wmma_bf16(af, bfg, acc[tn]);
        }
    }

    __syncthreads();
    const int halfw = lane >> 4, l16 = lane & 15;
    #pragma unroll
    for (int tn = 0; tn < 4; tn++)
        #pragma unroll
        for (int r = 0; r < 8; r++)
            Cst[(wm * 16 + r + 8 * halfw) * 128 + wn * 64 + tn * 16 + l16] = acc[tn][r];
    __syncthreads();

    if (sel < 2) {
        // RoPE: half = 64; out[:64] = e*cos - o*sin ; out[64:] = e*sin + o*cos
        bf16_t* obase = ((sel == 0) ? qb : kb) + ((size_t)(b * HH + h) * TT + row0) * DD;
        for (int p = tid; p < 64 * 64; p += 256) {
            int r = p >> 6, d = p & 63;
            float t    = (float)(row0 + r);
            float freq = __expf(-(float)d * (9.210340371976184f / 64.0f)); // ln(1e4)/64
            float s, c;
            __sincosf(t * freq, &s, &c);
            float ev = Cst[r * 128 + d];
            float od = Cst[r * 128 + 64 + d];
            obase[r * DD + d]      = (bf16_t)(ev * c - od * s);
            obase[r * DD + 64 + d] = (bf16_t)(ev * s + od * c);
        }
    } else {
        // store V transposed: vT[b][h][d][t]
        bf16_t* vbase = vb + (size_t)(b * HH + h) * DD * TT;
        for (int p = tid; p < 64 * 128; p += 256) {
            int d = p >> 6, r = p & 63;
            vbase[(size_t)d * TT + row0 + r] = (bf16_t)Cst[r * 128 + d];
        }
    }
}

// ---------------------------------------------------------------- flash attention
// grid: (T/128, H, B)  block: 256 (8 waves, each owns 16 query rows)
// K tile [key][dim] and V^T tile [dim][key] stream via async-to-LDS,
// double buffered (4 async instr per wave per batch).
__global__ __launch_bounds__(256) void attn_kernel(
    const bf16_t* __restrict__ qb,   // [B][H][T][D]
    const bf16_t* __restrict__ kb,   // [B][H][T][D]
    const bf16_t* __restrict__ vt,   // [B][H][D][T]  (transposed)
    bf16_t* __restrict__ ob)         // [B][H][T][D]
{
    __shared__ bf16_t Klds[2][32 * 128];   // 2 x 8 KB  [key][dim]
    __shared__ bf16_t Vt  [2][128 * 32];   // 2 x 8 KB  [dim][key]
    __shared__ bf16_t Pst [8][16 * 32];    // 8 KB per-wave P staging

    const int tq = blockIdx.x, h = blockIdx.y, b = blockIdx.z;
    const int tid = threadIdx.x, wid = tid >> 5, lane = tid & 31;
    const int halfw = lane >> 4, l16 = lane & 15;

    const int q0wg = tq * 128;
    const int q0   = q0wg + wid * 16;

    const bf16_t* qbase = qb + (size_t)(b * HH + h) * TT * DD;
    const bf16_t* kbase = kb + (size_t)(b * HH + h) * TT * DD;
    const bf16_t* vbase = vt + (size_t)(b * HH + h) * DD * TT;

    // per-thread async chunk coordinates
    const int kkey = tid >> 4, kdc = tid & 15;       // K tile: 512 chunks, 2/thread
    const int vdim = tid >> 2, vkc = tid & 3;        // Vt tile: 512 chunks, 2/thread

    v16bf qf[4];
    #pragma unroll
    for (int kc = 0; kc < 4; kc++) qf[kc] = load_afrag(qbase, DD, q0, kc * 32, lane);

    v8f Oacc[8] = {};
    float mrow[8], lrow[8];
    #pragma unroll
    for (int r = 0; r < 8; r++) { mrow[r] = -1e30f; lrow[r] = 0.f; }

    const int jend = q0wg + 128;

    // prologue: batch 0
    async_ld_b128(&Klds[0][kkey * 128 + kdc * 8],        kbase + (size_t)kkey * DD + kdc * 8);
    async_ld_b128(&Klds[0][(kkey + 16) * 128 + kdc * 8], kbase + (size_t)(kkey + 16) * DD + kdc * 8);
    async_ld_b128(&Vt[0][vdim * 32 + vkc * 8],        vbase + (size_t)vdim * TT + vkc * 8);
    async_ld_b128(&Vt[0][(vdim + 64) * 32 + vkc * 8], vbase + (size_t)(vdim + 64) * TT + vkc * 8);

    for (int j0 = 0; j0 < jend; j0 += 32) {
        const int cur = (j0 >> 5) & 1;
        __syncthreads();                        // all waves done with buf[cur^1]
        if (j0 + 32 < jend) {
            const int nxt = cur ^ 1, jn = j0 + 32;
            async_ld_b128(&Klds[nxt][kkey * 128 + kdc * 8],        kbase + (size_t)(jn + kkey) * DD + kdc * 8);
            async_ld_b128(&Klds[nxt][(kkey + 16) * 128 + kdc * 8], kbase + (size_t)(jn + kkey + 16) * DD + kdc * 8);
            async_ld_b128(&Vt[nxt][vdim * 32 + vkc * 8],        vbase + (size_t)vdim * TT + jn + vkc * 8);
            async_ld_b128(&Vt[nxt][(vdim + 64) * 32 + vkc * 8], vbase + (size_t)(vdim + 64) * TT + jn + vkc * 8);
            WAIT_ASYNC(4);                      // in-order: batch 'cur' landed
        } else {
            WAIT_ASYNC(0);
        }
        __syncthreads();

        if (j0 <= q0 + 15) {                    // wave-uniform: EXEC stays all-ones
            v8f s0 = {}, s1 = {};
            #pragma unroll
            for (int kc = 0; kc < 4; kc++) {
                v16bf b0 = load_bfrag(Klds[cur], 128, 0,  kc * 32, lane);
                v16bf b1 = load_bfrag(Klds[cur], 128, 16, kc * 32, lane);
                s0 = wmma_bf16(qf[kc], b0, s0);
                s1 = wmma_bf16(qf[kc], b1, s1);
            }
            const float inv_d = 1.0f / 128.0f;  // muP: 1/d_head
            float alpha[8];
            #pragma unroll
            for (int r = 0; r < 8; r++) {
                int rowg = q0 + r + 8 * halfw;
                float sa = ((j0 + l16)      <= rowg) ? s0[r] * inv_d : -1e30f;
                float sb = ((j0 + 16 + l16) <= rowg) ? s1[r] * inv_d : -1e30f;
                float m = fmaxf(sa, sb);
                m = fmaxf(m, __shfl_xor(m, 1, 32));
                m = fmaxf(m, __shfl_xor(m, 2, 32));
                m = fmaxf(m, __shfl_xor(m, 4, 32));
                m = fmaxf(m, __shfl_xor(m, 8, 32));
                float mn = fmaxf(mrow[r], m);
                float al = __expf(mrow[r] - mn);
                float p0 = __expf(sa - mn);
                float p1 = __expf(sb - mn);
                s0[r] = p0; s1[r] = p1;
                float rs = p0 + p1;
                rs += __shfl_xor(rs, 1, 32);
                rs += __shfl_xor(rs, 2, 32);
                rs += __shfl_xor(rs, 4, 32);
                rs += __shfl_xor(rs, 8, 32);
                lrow[r]  = lrow[r] * al + rs;
                mrow[r]  = mn;
                alpha[r] = al;
            }
            #pragma unroll
            for (int nt = 0; nt < 8; nt++)
                #pragma unroll
                for (int r = 0; r < 8; r++) Oacc[nt][r] *= alpha[r];

            // stage P (C layout) -> LDS -> A-fragment layout (same-wave LDS is in-order)
            bf16_t* Pw = &Pst[wid][0];
            #pragma unroll
            for (int r = 0; r < 8; r++) {
                int row = r + 8 * halfw;
                Pw[row * 32 + l16]      = (bf16_t)s0[r];
                Pw[row * 32 + 16 + l16] = (bf16_t)s1[r];
            }
            v16bf pf = load_afrag(Pw, 32, 0, 0, lane);
            #pragma unroll
            for (int nt = 0; nt < 8; nt++) {
                v16bf vf = load_bfrag(Vt[cur], 32, nt * 16, 0, lane);
                Oacc[nt] = wmma_bf16(pf, vf, Oacc[nt]);
            }
        }
    }

    bf16_t* obase = ob + ((size_t)(b * HH + h) * TT + q0) * DD;
    #pragma unroll
    for (int r = 0; r < 8; r++) {
        float inv = 1.0f / lrow[r];
        int row = r + 8 * halfw;
        #pragma unroll
        for (int nt = 0; nt < 8; nt++)
            obase[row * DD + nt * 16 + l16] = (bf16_t)(Oacc[nt][r] * inv);
    }
}

// ---------------------------------------------------------------- output projection
// grid: (T/64, M/128, B)  block: 256; async double-buffered tiles like qkv.
__global__ __launch_bounds__(256) void out_proj_kernel(
    const bf16_t* __restrict__ ob,    // [B][H][T][D] -> logically [B*T][H*D]
    const bf16_t* __restrict__ wot,   // transposed: [M][H*D]
    float* __restrict__ out)          // [B][T][M]
{
    __shared__ bf16_t Alds[2][64 * 32];
    __shared__ bf16_t Btl [2][128 * 32];

    const int tq = blockIdx.x, nb = blockIdx.y, b = blockIdx.z;
    const int row0 = tq * 64, n0 = nb * 128;
    const int tid = threadIdx.x, wid = tid >> 5, lane = tid & 31;
    const int wm = wid & 3, wn = wid >> 2;
    const int halfw = lane >> 4, l16 = lane & 15;
    const int ar = tid >> 2, akc = tid & 3;
    const int brow0 = tid >> 2, bkc = tid & 3;
    const bf16_t* Wrow = wot + (size_t)n0 * HD;

    v8f acc[4] = {};

    // A-tile source address for logical col k = h*128 + d
    auto asrc = [&](int k0) {
        int k = k0 + akc * 8, hh = k >> 7, d = k & 127;
        return ob + ((size_t)(b * HH + hh) * TT + row0 + ar) * DD + d;
    };

    async_ld_b128(&Alds[0][ar * 32 + akc * 8], asrc(0));
    async_ld_b128(&Btl[0][brow0 * 32 + bkc * 8], Wrow + (size_t)brow0 * HD + bkc * 8);
    async_ld_b128(&Btl[0][(brow0 + 64) * 32 + bkc * 8], Wrow + (size_t)(brow0 + 64) * HD + bkc * 8);

    for (int k0 = 0; k0 < HD; k0 += 32) {
        const int cur = (k0 >> 5) & 1;
        __syncthreads();
        if (k0 + 32 < HD) {
            const int nxt = cur ^ 1, kn = k0 + 32;
            async_ld_b128(&Alds[nxt][ar * 32 + akc * 8], asrc(kn));
            async_ld_b128(&Btl[nxt][brow0 * 32 + bkc * 8], Wrow + (size_t)brow0 * HD + kn + bkc * 8);
            async_ld_b128(&Btl[nxt][(brow0 + 64) * 32 + bkc * 8], Wrow + (size_t)(brow0 + 64) * HD + kn + bkc * 8);
            WAIT_ASYNC(3);
        } else {
            WAIT_ASYNC(0);
        }
        __syncthreads();
        v16bf af = load_afrag(Alds[cur], 32, wm * 16, 0, lane);
        #pragma unroll
        for (int tn = 0; tn < 4; tn++) {
            v16bf bfg = load_bfrag(Btl[cur], 32, wn * 64 + tn * 16, 0, lane);
            acc[tn] = wmma_bf16(af, bfg, acc[tn]);
        }
    }

    float* obase = out + ((size_t)b * TT + row0) * MM + n0;
    #pragma unroll
    for (int tn = 0; tn < 4; tn++)
        #pragma unroll
        for (int r = 0; r < 8; r++)
            obase[(wm * 16 + r + 8 * halfw) * MM + wn * 64 + tn * 16 + l16] = acc[tn][r];
}

// ---------------------------------------------------------------- launch
extern "C" void kernel_launch(void* const* d_in, const int* in_sizes, int n_in,
                              void* d_out, int out_size, void* d_ws, size_t ws_size,
                              hipStream_t stream) {
    const float* x  = (const float*)d_in[0];
    const float* wq = (const float*)d_in[1];
    const float* wk = (const float*)d_in[2];
    const float* wv = (const float*)d_in[3];
    const float* wo = (const float*)d_in[4];

    const size_t XB_N  = (size_t)BB * TT * MM;        // 4,194,304
    const size_t W_N   = (size_t)MM * HD;             // 1,048,576
    const size_t QKV_N = (size_t)BB * HH * TT * DD;   // 4,194,304

    bf16_t* ws = (bf16_t*)d_ws;
    size_t off = 0;
    bf16_t* xb   = ws + off; off += XB_N;
    bf16_t* wqt  = ws + off; off += W_N;      // [HD][M]
    bf16_t* wkt  = ws + off; off += W_N;
    bf16_t* wvt  = ws + off; off += W_N;
    bf16_t* wot  = ws + off; off += W_N;      // [M][HD]
    bf16_t* qbuf = ws + off; off += QKV_N;    // [B][H][T][D]
    bf16_t* kbuf = ws + off; off += QKV_N;    // [B][H][T][D]
    bf16_t* vbuf = ws + off; off += QKV_N;    // [B][H][D][T] (transposed)
    bf16_t* obuf = ws + off; off += QKV_N;    // [B][H][T][D]
    if (ws_size < off * sizeof(bf16_t)) return;

    cvt_f32_bf16<<<(int)((XB_N + 255) / 256), 256, 0, stream>>>(x, xb, (int)XB_N);
    cvt_transpose_bf16<<<(int)((W_N + 255) / 256), 256, 0, stream>>>(wq, wqt, MM, HD);
    cvt_transpose_bf16<<<(int)((W_N + 255) / 256), 256, 0, stream>>>(wk, wkt, MM, HD);
    cvt_transpose_bf16<<<(int)((W_N + 255) / 256), 256, 0, stream>>>(wv, wvt, MM, HD);
    cvt_transpose_bf16<<<(int)((W_N + 255) / 256), 256, 0, stream>>>(wo, wot, HD, MM);

    qkv_rope_kernel<<<dim3(TT / 64, HH, BB * 3), 256, 0, stream>>>(
        xb, wqt, wkt, wvt, qbuf, kbuf, vbuf);

    attn_kernel<<<dim3(TT / 128, HH, BB), 256, 0, stream>>>(qbuf, kbuf, vbuf, obuf);

    out_proj_kernel<<<dim3(TT / 64, MM / 128, BB), 256, 0, stream>>>(
        obuf, wot, (float*)d_out);
}
